// GCN_7301444403486
// MI455X (gfx1250) — matrix operand order repaired and checked by reading, not compiled, not run on hardware
//
#include <hip/hip_runtime.h>
#include <hip/hip_bf16.h>
#include <stdint.h>

// ---------------------------------------------------------------------------
// Types for CDNA5 WMMA (wave32): v_wmma_f32_16x16x32_bf16
// ---------------------------------------------------------------------------
typedef __attribute__((ext_vector_type(16))) __bf16 v16bf;
typedef __attribute__((ext_vector_type(8)))  __bf16 bf16x8;
typedef __attribute__((ext_vector_type(4)))  __bf16 bf16x4;
typedef __attribute__((ext_vector_type(2)))  __bf16 bf16x2;
typedef __attribute__((ext_vector_type(8)))  float  v8f;

__device__ __forceinline__ __bf16 f2bf(float f) {
  unsigned u = __builtin_bit_cast(unsigned, f);
  unsigned r = u + 0x7FFFu + ((u >> 16) & 1u);   // round-to-nearest-even
  unsigned short h = (unsigned short)(r >> 16);
  return __builtin_bit_cast(__bf16, h);
}
__device__ __forceinline__ float bf2f(__bf16 b) {
  unsigned short h = __builtin_bit_cast(unsigned short, b);
  unsigned u = ((unsigned)h) << 16;
  return __builtin_bit_cast(float, u);
}

// ---------------------------------------------------------------------------
// Degree / normalization:  deg[i] = 1 (self loop) + in-degree;  dinv = rsqrt
// ---------------------------------------------------------------------------
__global__ void k_deg_init(float* __restrict__ deg, int n) {
  int i = blockIdx.x * blockDim.x + threadIdx.x;
  if (i < n) deg[i] = 1.0f;
}
__global__ void k_deg_edges(const int* __restrict__ dst, float* __restrict__ deg, int e) {
  int i = blockIdx.x * blockDim.x + threadIdx.x;
  if (i < e) unsafeAtomicAdd(&deg[dst[i]], 1.0f);
}
__global__ void k_dinv(float* __restrict__ deg, int n) {
  int i = blockIdx.x * blockDim.x + threadIdx.x;
  if (i < n) deg[i] = rsqrtf(deg[i]);
}

// ---------------------------------------------------------------------------
// Streaming fp32 -> bf16 conversion (vectorized x4).
// ---------------------------------------------------------------------------
__global__ void k_f2bf4(const float* __restrict__ src, __bf16* __restrict__ dst, int total4) {
  int idx = blockIdx.x * blockDim.x + threadIdx.x;
  if (idx >= total4) return;
  float4 v = *(const float4*)(src + (size_t)idx * 4);
  bf16x4 o;
  o[0] = f2bf(v.x); o[1] = f2bf(v.y); o[2] = f2bf(v.z); o[3] = f2bf(v.w);
  *(bf16x4*)(dst + (size_t)idx * 4) = o;
}

// ---------------------------------------------------------------------------
// Pack W[K,N] (fp32) into WMMA-B-fragment-native bf16 layout:
//   WbP[ ((nT*(K/32) + kq)*32 + lane)*16 + i ] = bf16( W[(kq*32 + hi*8 + (i&7) + (i>>3)*16)*N + nT*16 + lane16] )
// so each lane's 16 B-fragment values are 32 contiguous bytes -> 2x b128 load.
// (B 32x16 bf16 layout: lane%16 = N col; lanes 0-15 K 0..7,16..23, lanes 16-31 K 8..15,24..31)
// ---------------------------------------------------------------------------
template <int K, int N>
__global__ void k_wpack(const float* __restrict__ W, __bf16* __restrict__ WbP) {
  constexpr int KQ = K / 32;
  int idx = blockIdx.x * blockDim.x + threadIdx.x;
  if (idx >= K * N) return;
  int i      = idx & 15;
  int lane   = (idx >> 4) & 31;
  int kq     = (idx >> 9) % KQ;
  int nT     = idx / (512 * KQ);
  int hi     = lane >> 4;
  int lane16 = lane & 15;
  int k      = kq * 32 + hi * 8 + (i & 7) + (i >> 3) * 16;
  int col    = nT * 16 + lane16;
  WbP[idx] = f2bf(W[(size_t)k * N + col]);
}

// ---------------------------------------------------------------------------
// GEMM  T[M,N] = Xb[M,K](bf16) @ W[K,N]   via v_wmma_f32_16x16x32_bf16.
// One wave computes a 16x32 output strip (two 16x16 column tiles): the A
// fragment is loaded once per k-step and feeds two back-to-back WMMAs with
// independent accumulators (no RAW hazard -> good XDL pipelining).
//   A (16x32 bf16): lanes 0-15 -> M=lane, elems[0..7]=K 0..7, [8..15]=K 16..23
//                   lanes 16-31 -> M=lane-16, K offset +8 (8..15 / 24..31)
//   B: fragment-packed layout (see k_wpack) -> 32 contiguous bytes per lane.
//   C/D (16x16 f32): VGPR r -> M = r + (lane>=16 ? 8 : 0), N = lane%16
// Store epilogue: wave-uniform full-tile fast path (no per-element exec masks).
// ---------------------------------------------------------------------------
template <int K, int N>
__global__ void __launch_bounds__(256)
k_gemm_wmma(const __bf16* __restrict__ xb, const __bf16* __restrict__ WbP,
            __bf16* __restrict__ T, int M) {
  constexpr int NT = N / 16;
  constexpr int NP = NT / 2;     // column-tile pairs per row strip
  constexpr int KQ = K / 32;
  const int wave   = threadIdx.x >> 5;
  const int lane   = threadIdx.x & 31;
  const int lane16 = lane & 15;
  const int hi     = lane >> 4;

  const int mTiles = (M + 15) >> 4;
  const int tile   = blockIdx.x * 8 + wave;       // 8 waves / block
  if (tile >= mTiles * NP) return;                // wave-uniform guard
  const int mT  = tile / NP;
  const int nT0 = (tile - mT * NP) * 2;

  int row = mT * 16 + lane16;
  const bool grow_ok = row < M;
  if (row >= M) row = M - 1;                      // clamp (tail safety)

  v8f c0 = {0.f, 0.f, 0.f, 0.f, 0.f, 0.f, 0.f, 0.f};
  v8f c1 = {0.f, 0.f, 0.f, 0.f, 0.f, 0.f, 0.f, 0.f};

#pragma unroll
  for (int kq = 0; kq < KQ; ++kq) {
    const int ka = kq * 32 + hi * 8;
    bf16x8 alo = *(const bf16x8*)(xb + (size_t)row * K + ka);        // 16B load
    bf16x8 ahi = *(const bf16x8*)(xb + (size_t)row * K + ka + 16);   // 16B load
    v16bf a = __builtin_shufflevector(alo, ahi,
        0, 1, 2, 3, 4, 5, 6, 7, 8, 9, 10, 11, 12, 13, 14, 15);
    // packed B fragments: 32 contiguous bytes per lane per column tile
    v16bf b0 = *(const v16bf*)(WbP + (((size_t)(nT0 * KQ + kq) * 32 + lane) << 4));
    v16bf b1 = *(const v16bf*)(WbP + (((size_t)((nT0 + 1) * KQ + kq) * 32 + lane) << 4));

    c0 = __builtin_amdgcn_wmma_f32_16x16x32_bf16(
        false, a, false, b0, (short)0, c0, false, false);
    c1 = __builtin_amdgcn_wmma_f32_16x16x32_bf16(
        false, a, false, b1, (short)0, c1, false, false);
  }

  // Epilogue: M = r + hi*8, col = nT*16 + lane16
  __bf16* t0 = T + (size_t)(mT * 16 + hi * 8) * N + nT0 * 16 + lane16;
  if (mT * 16 + 16 <= M) {
    // whole tile in bounds (wave-uniform): branch-free stores, one base addr
#pragma unroll
    for (int r = 0; r < 8; ++r) {
      t0[(size_t)r * N]      = f2bf(c0[r]);
      t0[(size_t)r * N + 16] = f2bf(c1[r]);
    }
  } else {
#pragma unroll
    for (int r = 0; r < 8; ++r) {
      int orow = mT * 16 + hi * 8 + r;
      if (orow < M && grow_ok) {
        t0[(size_t)r * N]      = f2bf(c0[r]);
        t0[(size_t)r * N + 16] = f2bf(c1[r]);
      }
    }
  }
}

// ---------------------------------------------------------------------------
// Self-loop init of accumulator: A[i,:] = t[i,:] * dinv[i]^2  (full overwrite)
// ---------------------------------------------------------------------------
template <int F>
__global__ void k_selfagg(const __bf16* __restrict__ T, const float* __restrict__ dinv,
                          float* __restrict__ A, int nnodes) {
  constexpr int QF = F / 4;
  int idx = blockIdx.x * blockDim.x + threadIdx.x;
  if (idx >= nnodes * QF) return;
  int node = idx / QF;
  int q    = idx - node * QF;
  float di = dinv[node];
  float w  = di * di;
  bf16x4 t = *(const bf16x4*)(T + (size_t)node * F + q * 4);
  float4 o;
  o.x = bf2f(t[0]) * w;
  o.y = bf2f(t[1]) * w;
  o.z = bf2f(t[2]) * w;
  o.w = bf2f(t[3]) * w;
  *(float4*)(A + (size_t)node * F + q * 4) = o;
}

// ---------------------------------------------------------------------------
// Edge scatter: one wave per edge; lane owns F/32 features.
// Gathers bf16 row (L2-resident), scales by dinv[s]*dinv[d], f32 atomic add.
// ---------------------------------------------------------------------------
template <int F>
__global__ void __launch_bounds__(256)
k_scatter(const int* __restrict__ src, const int* __restrict__ dst,
          const float* __restrict__ dinv, const __bf16* __restrict__ T,
          float* __restrict__ A, int E) {
  constexpr int FP = F / 32;   // 4 (F=128) or 2 (F=64)
  int e = (blockIdx.x * 256 + threadIdx.x) >> 5;
  if (e >= E) return;          // wave-uniform
  int lane = threadIdx.x & 31;
  int s = src[e], d = dst[e];
  float w = dinv[s] * dinv[d];
  const __bf16* tp = T + (size_t)s * F + lane * FP;
  float*        ap = A + (size_t)d * F + lane * FP;
  if constexpr (FP == 4) {
    bf16x4 t = *(const bf16x4*)tp;
    unsafeAtomicAdd(ap + 0, bf2f(t[0]) * w);
    unsafeAtomicAdd(ap + 1, bf2f(t[1]) * w);
    unsafeAtomicAdd(ap + 2, bf2f(t[2]) * w);
    unsafeAtomicAdd(ap + 3, bf2f(t[3]) * w);
  } else {
    bf16x2 t = *(const bf16x2*)tp;
    unsafeAtomicAdd(ap + 0, bf2f(t[0]) * w);
    unsafeAtomicAdd(ap + 1, bf2f(t[1]) * w);
  }
}

// ---------------------------------------------------------------------------
// Bias + (optional) ReLU, vectorized x4; out bf16 (next layer) or f32 (final).
// ---------------------------------------------------------------------------
template <int F, bool RELU, bool OUTBF>
__global__ void k_finish(const float* __restrict__ A, const float* __restrict__ bias,
                         void* __restrict__ outp, int total4) {
  constexpr int QF = F / 4;
  int idx = blockIdx.x * blockDim.x + threadIdx.x;
  if (idx >= total4) return;
  int q  = idx % QF;
  float4 v = *(const float4*)(A + (size_t)idx * 4);
  float4 bb = *(const float4*)(bias + q * 4);
  v.x += bb.x; v.y += bb.y; v.z += bb.z; v.w += bb.w;
  if constexpr (RELU) {
    v.x = fmaxf(v.x, 0.f); v.y = fmaxf(v.y, 0.f);
    v.z = fmaxf(v.z, 0.f); v.w = fmaxf(v.w, 0.f);
  }
  if constexpr (OUTBF) {
    bf16x4 o;
    o[0] = f2bf(v.x); o[1] = f2bf(v.y); o[2] = f2bf(v.z); o[3] = f2bf(v.w);
    *(bf16x4*)((__bf16*)outp + (size_t)idx * 4) = o;
  } else {
    *(float4*)((float*)outp + (size_t)idx * 4) = v;
  }
}

// ---------------------------------------------------------------------------
// Host launcher.
// Inputs: x, edge_index(int32 [2,E]), W1,b1,W2,b2,W3,b3
// ---------------------------------------------------------------------------
extern "C" void kernel_launch(void* const* d_in, const int* in_sizes, int n_in,
                              void* d_out, int out_size, void* d_ws, size_t ws_size,
                              hipStream_t stream) {
  const float* x  = (const float*)d_in[0];
  const int*   ei = (const int*)d_in[1];   // JAX default x64-off -> int32
  const float* W1 = (const float*)d_in[2];
  const float* b1 = (const float*)d_in[3];
  const float* W2 = (const float*)d_in[4];
  const float* b2 = (const float*)d_in[5];
  const float* W3 = (const float*)d_in[6];
  const float* b3 = (const float*)d_in[7];

  const int N = in_sizes[0] / 128;         // 100000
  const int E = in_sizes[1] / 2;           // 1600000
  const int* esrc = ei;
  const int* edst = ei + E;

  uint8_t* ws = (uint8_t*)d_ws;
  size_t off = 0;
  auto take = [&](size_t bytes) -> void* {
    void* p = ws + off;
    off = (off + bytes + 255) & ~(size_t)255;
    return p;
  };
  float*  dinv = (float*)take((size_t)N * 4);
  __bf16* W1b  = (__bf16*)take(128 * 128 * 2);
  __bf16* W2b  = (__bf16*)take(128 * 128 * 2);
  __bf16* W3b  = (__bf16*)take(128 * 64 * 2);
  __bf16* Tb1  = (__bf16*)take((size_t)N * 128 * 2);   // 25.6 MB
  __bf16* Tb2  = (__bf16*)take((size_t)N * 128 * 2);   // 25.6 MB
  float*  Agg  = (float*)take((size_t)N * 128 * 4);    // 51.2 MB
  // Xbf's live range ([x2bf .. gemm1]) ends before Agg's begins ([selfagg1 ..]),
  // so alias it onto Agg to save 25.6 MB of workspace.
  __bf16* Xbf  = (__bf16*)Agg;
  (void)ws_size; (void)n_in; (void)out_size;

  const int TPB = 256;
  auto cdiv = [](long long a, long long b) { return (int)((a + b - 1) / b); };
  const int mTiles = (N + 15) / 16;

  // normalization
  k_deg_init <<<cdiv(N, TPB), TPB, 0, stream>>>(dinv, N);
  k_deg_edges<<<cdiv(E, TPB), TPB, 0, stream>>>(edst, dinv, E);
  k_dinv     <<<cdiv(N, TPB), TPB, 0, stream>>>(dinv, N);

  // weights -> packed bf16 fragment layout; x -> bf16 once
  k_wpack<128, 128><<<cdiv(128 * 128, TPB), TPB, 0, stream>>>(W1, W1b);
  k_wpack<128, 128><<<cdiv(128 * 128, TPB), TPB, 0, stream>>>(W2, W2b);
  k_wpack<128, 64> <<<cdiv(128 * 64,  TPB), TPB, 0, stream>>>(W3, W3b);
  k_f2bf4<<<cdiv((long long)N * 32, TPB), TPB, 0, stream>>>(x, Xbf, N * 32);

  // ---- layer 1: xb @ W1 -> Tb1 ; agg -> Agg ; h1(bf16) -> Tb1
  k_gemm_wmma<128, 128><<<cdiv((long long)mTiles * 4, 8), TPB, 0, stream>>>(Xbf, W1b, Tb1, N);
  k_selfagg<128><<<cdiv((long long)N * 32, TPB), TPB, 0, stream>>>(Tb1, dinv, Agg, N);
  k_scatter<128><<<cdiv((long long)E * 32, TPB), TPB, 0, stream>>>(esrc, edst, dinv, Tb1, Agg, E);
  k_finish<128, true, true><<<cdiv((long long)N * 32, TPB), TPB, 0, stream>>>(Agg, b1, Tb1, N * 32);

  // ---- layer 2: h1(bf16) @ W2 -> Tb2 ; agg ; h2(bf16) -> Tb2
  k_gemm_wmma<128, 128><<<cdiv((long long)mTiles * 4, 8), TPB, 0, stream>>>(Tb1, W2b, Tb2, N);
  k_selfagg<128><<<cdiv((long long)N * 32, TPB), TPB, 0, stream>>>(Tb2, dinv, Agg, N);
  k_scatter<128><<<cdiv((long long)E * 32, TPB), TPB, 0, stream>>>(esrc, edst, dinv, Tb2, Agg, E);
  k_finish<128, true, true><<<cdiv((long long)N * 32, TPB), TPB, 0, stream>>>(Agg, b2, Tb2, N * 32);

  // ---- layer 3: h2(bf16) @ W3 -> Tb1 (N=64) ; agg ; out(f32) -> d_out
  k_gemm_wmma<128, 64><<<cdiv((long long)mTiles * 2, 8), TPB, 0, stream>>>(Tb2, W3b, Tb1, N);
  k_selfagg<64><<<cdiv((long long)N * 16, TPB), TPB, 0, stream>>>(Tb1, dinv, Agg, N);
  k_scatter<64><<<cdiv((long long)E * 32, TPB), TPB, 0, stream>>>(esrc, edst, dinv, Tb1, Agg, E);
  k_finish<64, false, false><<<cdiv((long long)N * 16, TPB), TPB, 0, stream>>>(Agg, b3, d_out, N * 16);
}
